// Attention_28948079575308
// MI455X (gfx1250) — matrix operand order repaired and checked by reading, not compiled, not run on hardware
//
#include <hip/hip_runtime.h>
#include <hip/hip_bf16.h>

// ---------------- problem constants ----------------
#define BB      2
#define SDIM    2048
#define HDS     16
#define DMODEL  2048
#define QLORA   1536
#define KVLORA  512
#define KVAN    576          // KV_LORA + ROPE
#define NOPE_   128
#define ROPE_   64
#define VD      128
#define QKH     192          // NOPE + ROPE
#define QOUT    3072         // HDS*QKH
#define KVBN    4096         // HDS*(NOPE+VD)
#define SCALE_  0.07216878364870322f   // 192^-0.5
#define EPS_    1e-6f

typedef __attribute__((ext_vector_type(16))) __bf16 v16bf;
typedef __attribute__((ext_vector_type(8)))  float  v8f;
typedef __attribute__((ext_vector_type(4)))  int    v4i_;

// ---------------- async global->LDS copy (CDNA5), with portable fallback ----------------
#if defined(__has_builtin)
#if __has_builtin(__builtin_amdgcn_global_load_async_to_lds_b128)
#define HAVE_ASYNC_LDS 1
#endif
#endif
#ifndef HAVE_ASYNC_LDS
#define HAVE_ASYNC_LDS 0
#endif

__device__ __forceinline__ void cp16(void* l, const void* g) {
#if HAVE_ASYNC_LDS
  __builtin_amdgcn_global_load_async_to_lds_b128(
      (__attribute__((address_space(1))) v4i_*)g,
      (__attribute__((address_space(3))) v4i_*)l, 0, 0);
#else
  *(uint4*)l = *(const uint4*)g;
#endif
}

__device__ __forceinline__ void cp_wait() {
#if HAVE_ASYNC_LDS
#if __has_builtin(__builtin_amdgcn_s_wait_asynccnt)
  __builtin_amdgcn_s_wait_asynccnt(0);
#else
  asm volatile("s_wait_asynccnt 0" ::: "memory");
#endif
#endif
}

// ---------------- WMMA fragment helpers (CDNA5 wave32 layouts) ----------------
// A-matrix 16x32 bf16: lanes 0-15 -> M=lane, K={0..7,16..23}; lanes 16-31 -> M=lane-16, K={8..15,24..31}
__device__ __forceinline__ v16bf load_a_frag(const __bf16* base, int ld, int lane) {
  v16bf a;
  int row = lane & 15;
  int hk  = (lane >> 4) * 8;
  const __bf16* p = base + row * ld;
#pragma unroll
  for (int i = 0; i < 8; ++i) a[i]     = p[hk + i];
#pragma unroll
  for (int i = 0; i < 8; ++i) a[8 + i] = p[16 + hk + i];
  return a;
}

// B-matrix 32x16 with B[k][n] = base[n*ld + k]; lanes 0-15: K={0..7,16..23}, lanes 16-31: K={8..15,24..31}
__device__ __forceinline__ v16bf load_bt_frag(const __bf16* base, int ld, int lane) {
  v16bf b;
  int col = lane & 15;
  int kb  = (lane >> 4) * 8;
  const __bf16* p = base + col * ld;
#pragma unroll
  for (int i = 0; i < 8; ++i) { b[i] = p[kb + i]; b[8 + i] = p[16 + kb + i]; }
  return b;
}

__device__ __forceinline__ v8f wmma_bf16(v16bf a, v16bf b, v8f c) {
  return __builtin_amdgcn_wmma_f32_16x16x32_bf16(false, a, false, b, (short)0, c, false, false);
}

// ---------------- f32 -> bf16 bulk convert (n divisible by 1024) ----------------
__global__ __launch_bounds__(256) void cvt_bf16(const float* __restrict__ s,
                                                __bf16* __restrict__ d, int n) {
  int i = (blockIdx.x * 256 + threadIdx.x) * 4;
  float4 f = *(const float4*)(s + i);
  __bf16* dp = d + i;
  dp[0] = (__bf16)f.x; dp[1] = (__bf16)f.y; dp[2] = (__bf16)f.z; dp[3] = (__bf16)f.w;
}

// ---------------- GEMM: C[M,N] = A[M,K] * W[N,K]^T, bf16 in, f32 out ----------------
// block 256 thr = 8 waves as 2(M) x 4(N); wave tile 32x64 => block tile 64 x 256; K-step 32.
__global__ __launch_bounds__(256) void gemm_bf16(const __bf16* __restrict__ A,
                                                 const __bf16* __restrict__ W,
                                                 float* __restrict__ C,
                                                 int M, int N, int K) {
  __shared__ __bf16 As[64][32];    // 4 KB
  __shared__ __bf16 Ws[256][32];   // 16 KB
  const int tid  = threadIdx.x;
  const int lane = tid & 31;
  const int w    = tid >> 5;
  const int wm   = w & 1;          // 0..1 -> 32-row sub-tile
  const int wn   = w >> 1;         // 0..3 -> 64-col sub-tile
  const int mb   = blockIdx.y * 64;
  const int nb   = blockIdx.x * 256;

  v8f acc[2][4];
#pragma unroll
  for (int i = 0; i < 2; ++i)
#pragma unroll
    for (int j = 0; j < 4; ++j) acc[i][j] = (v8f){};

  for (int k0 = 0; k0 < K; k0 += 32) {
    __syncthreads();
    {
      int idx = tid * 8;                     // 2048 elems = 256 thr * 8
      int r = idx >> 5, c = idx & 31;
      cp16(&As[r][c], A + (size_t)(mb + r) * K + k0 + c);
    }
#pragma unroll
    for (int it = 0; it < 4; ++it) {         // 8192 elems
      int idx = it * 2048 + tid * 8;
      int r = idx >> 5, c = idx & 31;
      if (nb + r < N)
        cp16(&Ws[r][c], W + (size_t)(nb + r) * K + k0 + c);
    }
    cp_wait();
    __syncthreads();

    v16bf a0 = load_a_frag(&As[wm * 32][0], 32, lane);
    v16bf a1 = load_a_frag(&As[wm * 32 + 16][0], 32, lane);
#pragma unroll
    for (int j = 0; j < 4; ++j) {
      v16bf b = load_bt_frag(&Ws[wn * 64 + j * 16][0], 32, lane);
      acc[0][j] = wmma_bf16(a0, b, acc[0][j]);
      acc[1][j] = wmma_bf16(a1, b, acc[1][j]);
    }
  }

#pragma unroll
  for (int i = 0; i < 2; ++i) {
    int rbase = mb + wm * 32 + i * 16 + ((lane >> 4) << 3);
#pragma unroll
    for (int j = 0; j < 4; ++j) {
      int col = nb + wn * 64 + j * 16 + (lane & 15);
      if (col < N) {
#pragma unroll
        for (int v = 0; v < 8; ++v)
          C[(size_t)(rbase + v) * N + col] = acc[i][j][v];
      }
    }
  }
}

// ---------------- RMSNorm over D leading elements of each row (f32 in, bf16 out) -------------
__global__ __launch_bounds__(256) void rmsnorm_rows(const float* __restrict__ X, int ldx, int D,
                                                    const float* __restrict__ Wt,
                                                    __bf16* __restrict__ Y) {
  __shared__ float red[256];
  const int row = blockIdx.x;
  const float* x = X + (size_t)row * ldx;
  __bf16* y = Y + (size_t)row * D;
  float s = 0.f;
  for (int i = threadIdx.x; i < D; i += 256) { float v = x[i]; s += v * v; }
  red[threadIdx.x] = s;
  __syncthreads();
  for (int o = 128; o > 0; o >>= 1) {
    if (threadIdx.x < o) red[threadIdx.x] += red[threadIdx.x + o];
    __syncthreads();
  }
  const float inv = rsqrtf(red[0] / (float)D + EPS_);
  for (int i = threadIdx.x; i < D; i += 256) y[i] = (__bf16)(x[i] * inv * Wt[i]);
}

// ---------------- RoPE + pack: Q/K [b][h][s][d], V transposed [b][h][d][s] ----------------
__global__ __launch_bounds__(256) void rope_pack(const float* __restrict__ qproj,  // [B*S, 3072]
                                                 const float* __restrict__ kvp,    // [B*S, 576]
                                                 const float* __restrict__ kvbp,   // [B*S, 4096]
                                                 const float* __restrict__ cosT,   // [S, 32]
                                                 const float* __restrict__ sinT,   // [S, 32]
                                                 __bf16* __restrict__ Qb,
                                                 __bf16* __restrict__ Kb,
                                                 __bf16* __restrict__ Vb) {
  const int bs = blockIdx.x;          // b*S + s
  const int s  = bs % SDIM;
  const int b  = bs / SDIM;

  for (int i = threadIdx.x; i < HDS * QKH; i += 256) {
    int h = i / QKH, d = i % QKH;
    float val;
    const float* qh = qproj + (size_t)bs * QOUT + h * QKH;
    if (d < NOPE_) {
      val = qh[d];
    } else {
      int j = (d - NOPE_) >> 1, odd = (d - NOPE_) & 1;
      float x0 = qh[NOPE_ + 2 * j], x1 = qh[NOPE_ + 2 * j + 1];
      float c = cosT[s * 32 + j], sn = sinT[s * 32 + j];
      val = odd ? (x0 * sn + x1 * c) : (x0 * c - x1 * sn);
    }
    Qb[(((size_t)(b * HDS + h)) * SDIM + s) * QKH + d] = (__bf16)(val * SCALE_);
  }
  for (int i = threadIdx.x; i < HDS * QKH; i += 256) {
    int h = i / QKH, d = i % QKH;
    float val;
    if (d < NOPE_) {
      val = kvbp[(size_t)bs * KVBN + h * (NOPE_ + VD) + d];
    } else {
      int j = (d - NOPE_) >> 1, odd = (d - NOPE_) & 1;
      const float* kp = kvp + (size_t)bs * KVAN + KVLORA;
      float x0 = kp[2 * j], x1 = kp[2 * j + 1];
      float c = cosT[s * 32 + j], sn = sinT[s * 32 + j];
      val = odd ? (x0 * sn + x1 * c) : (x0 * c - x1 * sn);
    }
    Kb[(((size_t)(b * HDS + h)) * SDIM + s) * QKH + d] = (__bf16)val;
  }
  // V stored d-major so attention tiles are contiguous copies
  for (int i = threadIdx.x; i < HDS * VD; i += 256) {
    int h = i / VD, d = i % VD;
    Vb[(((size_t)(b * HDS + h)) * VD + d) * SDIM + s] =
        (__bf16)kvbp[(size_t)bs * KVBN + h * (NOPE_ + VD) + NOPE_ + d];
  }
}

// ---------------- causal flash attention ----------------
__device__ __forceinline__ float half16_max(float v) {
#pragma unroll
  for (int m = 1; m <= 8; m <<= 1) v = fmaxf(v, __shfl_xor(v, m, 32));
  return v;
}
__device__ __forceinline__ float half16_sum(float v) {
#pragma unroll
  for (int m = 1; m <= 8; m <<= 1) v += __shfl_xor(v, m, 32);
  return v;
}

#define QTILE 128
// grid: (S/QTILE, B*HDS); block 256 = 8 waves; each wave owns 16 query rows.
__global__ __launch_bounds__(256) void mla_attn(const __bf16* __restrict__ Qb,
                                                const __bf16* __restrict__ Kb,
                                                const __bf16* __restrict__ Vb,   // [b][h][VD][S]
                                                __bf16* __restrict__ Out) {      // [B*S, HDS*VD]
  __shared__ __bf16 Ks[32][QKH];      // 12 KB
  __shared__ __bf16 Vst[VD][32];      // 8 KB  (d-major: rows = v-dim, cols = keys)
  __shared__ __bf16 Ps[8][16][32];    // 8 KB  (per-wave P staging)

  const int bh   = blockIdx.y;
  const int b    = bh / HDS;
  const int h    = bh % HDS;
  const int q0   = blockIdx.x * QTILE;
  const int tid  = threadIdx.x;
  const int lane = tid & 31;
  const int w    = tid >> 5;

  const __bf16* Qp = Qb + ((size_t)bh * SDIM + q0 + w * 16) * QKH;
  const __bf16* Kp = Kb + (size_t)bh * SDIM * QKH;
  const __bf16* Vp = Vb + (size_t)bh * VD * SDIM;

  // Q fragments for the 6 d-chunks of 32
  v16bf qa[6];
  {
    int row = lane & 15, hk = (lane >> 4) * 8;
    const __bf16* p = Qp + row * QKH;
#pragma unroll
    for (int dc = 0; dc < 6; ++dc) {
#pragma unroll
      for (int i = 0; i < 8; ++i) qa[dc][i]     = p[dc * 32 + hk + i];
#pragma unroll
      for (int i = 0; i < 8; ++i) qa[dc][8 + i] = p[dc * 32 + 16 + hk + i];
    }
  }

  float mrow[8], lrow[8];
#pragma unroll
  for (int v = 0; v < 8; ++v) { mrow[v] = -3.0e38f; lrow[v] = 0.f; }
  v8f oacc[8];
#pragma unroll
  for (int n = 0; n < 8; ++n) oacc[n] = (v8f){};

  const int rbase = q0 + w * 16 + ((lane >> 4) << 3);
  const int kend  = q0 + QTILE;

  for (int kt = 0; kt < kend; kt += 32) {
    __syncthreads();
#pragma unroll
    for (int it = 0; it < 3; ++it) {               // K tile: 32x192 = 6144 bf16
      int idx = it * 2048 + tid * 8;
      int r = idx / QKH, c = idx % QKH;
      cp16(&Ks[r][c], Kp + (size_t)(kt + r) * QKH + c);
    }
#pragma unroll
    for (int it = 0; it < 2; ++it) {               // V tile: 128x32 = 4096 bf16 (d-major)
      int idx = it * 2048 + tid * 8;
      int d = idx >> 5, t = idx & 31;
      cp16(&Vst[d][t], Vp + (size_t)d * SDIM + kt + t);
    }
    if (kt + 32 < kend) __builtin_prefetch(Kp + (size_t)(kt + 32) * QKH, 0, 0);
    cp_wait();
    __syncthreads();

    // scores for two 16-key subtiles
    v8f sc[2];
#pragma unroll
    for (int sub = 0; sub < 2; ++sub) {
      v8f s = {};
      const int col = lane & 15, kb0 = (lane >> 4) * 8;
#pragma unroll
      for (int dc = 0; dc < 6; ++dc) {
        v16bf kb;
        const __bf16* p = &Ks[sub * 16 + col][dc * 32];
#pragma unroll
        for (int i = 0; i < 8; ++i) { kb[i] = p[kb0 + i]; kb[8 + i] = p[16 + kb0 + i]; }
        s = wmma_bf16(qa[dc], kb, s);
      }
#pragma unroll
      for (int v = 0; v < 8; ++v) {
        int trow = rbase + v;
        int tcol = kt + sub * 16 + col;
        if (tcol > trow) s[v] = -3.0e38f;
      }
      sc[sub] = s;
    }

    // online softmax per row (rows replicated across the 16 lanes of each half)
#pragma unroll
    for (int v = 0; v < 8; ++v) {
      float tmax = half16_max(fmaxf(sc[0][v], sc[1][v]));
      float mnew = fmaxf(mrow[v], tmax);
      float alpha = __expf(mrow[v] - mnew);
      float p0 = __expf(sc[0][v] - mnew);
      float p1 = __expf(sc[1][v] - mnew);
      float tsum = half16_sum(p0 + p1);
      lrow[v] = lrow[v] * alpha + tsum;
      mrow[v] = mnew;
#pragma unroll
      for (int n = 0; n < 8; ++n) oacc[n][v] *= alpha;
      int prow = ((lane >> 4) << 3) + v;
      Ps[w][prow][lane & 15]        = (__bf16)p0;
      Ps[w][prow][16 + (lane & 15)] = (__bf16)p1;
    }
    asm volatile("s_wait_dscnt 0" ::: "memory");  // per-wave P write -> read

    // O += P (16x32) * V (32x128)  [V^T tile is d-major => contiguous B fragments]
    v16bf pa = load_a_frag(&Ps[w][0][0], 32, lane);
#pragma unroll
    for (int n = 0; n < 8; ++n) {
      v16bf vb = load_bt_frag(&Vst[n * 16][0], 32, lane);
      oacc[n] = wmma_bf16(pa, vb, oacc[n]);
    }
  }

  // normalize + write bf16 [B*S, HDS*VD]
#pragma unroll
  for (int n = 0; n < 8; ++n) {
    int col = h * VD + n * 16 + (lane & 15);
#pragma unroll
    for (int v = 0; v < 8; ++v) {
      int row = rbase + v;
      Out[(size_t)(b * SDIM + row) * (HDS * VD) + col] = (__bf16)(oacc[n][v] / lrow[v]);
    }
  }
}

// ---------------- host orchestration ----------------
extern "C" void kernel_launch(void* const* d_in, const int* in_sizes, int n_in,
                              void* d_out, int out_size, void* d_ws, size_t ws_size,
                              hipStream_t stream) {
  (void)in_sizes; (void)n_in; (void)out_size; (void)ws_size;
  const float* x        = (const float*)d_in[0];
  const float* fcos     = (const float*)d_in[1];
  const float* fsin     = (const float*)d_in[2];
  const float* wq_a     = (const float*)d_in[3];
  const float* q_norm_w = (const float*)d_in[4];
  const float* wq_b     = (const float*)d_in[5];
  const float* wkv_a    = (const float*)d_in[6];
  const float* kv_norm_w= (const float*)d_in[7];
  const float* wkv_b    = (const float*)d_in[8];
  const float* wo       = (const float*)d_in[9];
  float* out = (float*)d_out;

  const int M = BB * SDIM;  // 4096 token rows
  char* ws = (char*)d_ws;
  size_t off = 0;
  auto carve = [&](size_t bytes) -> void* {
    void* p = ws + off;
    off += (bytes + 255) & ~(size_t)255;
    return p;
  };
  // bf16 operand copies
  __bf16* xb    = (__bf16*)carve((size_t)M * DMODEL * 2);
  __bf16* wqab  = (__bf16*)carve((size_t)QLORA * DMODEL * 2);
  __bf16* wqbb  = (__bf16*)carve((size_t)QOUT * QLORA * 2);
  __bf16* wkvab = (__bf16*)carve((size_t)KVAN * DMODEL * 2);
  __bf16* wkvbb = (__bf16*)carve((size_t)KVBN * KVLORA * 2);
  __bf16* wob   = (__bf16*)carve((size_t)DMODEL * DMODEL * 2);
  // activations
  float*  q_lat = (float*)carve((size_t)M * QLORA * 4);
  __bf16* qn    = (__bf16*)carve((size_t)M * QLORA * 2);
  float*  kvp   = (float*)carve((size_t)M * KVAN * 4);
  __bf16* kvn   = (__bf16*)carve((size_t)M * KVLORA * 2);
  float*  qproj = (float*)carve((size_t)M * QOUT * 4);
  float*  kvbp  = (float*)carve((size_t)M * KVBN * 4);
  __bf16* Qbuf  = (__bf16*)carve((size_t)M * HDS * QKH * 2);
  __bf16* Kbuf  = (__bf16*)carve((size_t)M * HDS * QKH * 2);
  __bf16* Vbuf  = (__bf16*)carve((size_t)M * HDS * VD * 2);
  __bf16* attnb = (__bf16*)carve((size_t)M * HDS * VD * 2);

  dim3 blk(256);
  // 0) one-time bf16 conversions (all sizes divisible by 1024)
  cvt_bf16<<<(M * DMODEL) / 1024, blk, 0, stream>>>(x, xb, M * DMODEL);
  cvt_bf16<<<(QLORA * DMODEL) / 1024, blk, 0, stream>>>(wq_a, wqab, QLORA * DMODEL);
  cvt_bf16<<<(QOUT * QLORA) / 1024, blk, 0, stream>>>(wq_b, wqbb, QOUT * QLORA);
  cvt_bf16<<<(KVAN * DMODEL) / 1024, blk, 0, stream>>>(wkv_a, wkvab, KVAN * DMODEL);
  cvt_bf16<<<(KVBN * KVLORA) / 1024, blk, 0, stream>>>(wkv_b, wkvbb, KVBN * KVLORA);
  cvt_bf16<<<(DMODEL * DMODEL) / 1024, blk, 0, stream>>>(wo, wob, DMODEL * DMODEL);

  // 1) low-rank projections from x
  gemm_bf16<<<dim3((QLORA + 255) / 256, M / 64), blk, 0, stream>>>(xb, wqab, q_lat, M, QLORA, DMODEL);
  gemm_bf16<<<dim3((KVAN + 255) / 256, M / 64), blk, 0, stream>>>(xb, wkvab, kvp, M, KVAN, DMODEL);
  // 2) RMSNorms (bf16 out)
  rmsnorm_rows<<<M, 256, 0, stream>>>(q_lat, QLORA, QLORA, q_norm_w, qn);
  rmsnorm_rows<<<M, 256, 0, stream>>>(kvp, KVAN, KVLORA, kv_norm_w, kvn);
  // 3) up-projections
  gemm_bf16<<<dim3(QOUT / 256, M / 64), blk, 0, stream>>>(qn, wqbb, qproj, M, QOUT, QLORA);
  gemm_bf16<<<dim3(KVBN / 256, M / 64), blk, 0, stream>>>(kvn, wkvbb, kvbp, M, KVBN, KVLORA);
  // 4) RoPE + bf16 pack (V transposed)
  rope_pack<<<M, 256, 0, stream>>>(qproj, kvp, kvbp, fcos, fsin, Qbuf, Kbuf, Vbuf);
  // 5) causal flash attention (bf16 out)
  mla_attn<<<dim3(SDIM / QTILE, BB * HDS), 256, 0, stream>>>(Qbuf, Kbuf, Vbuf, attnb);
  // 6) output projection -> f32 d_out
  gemm_bf16<<<dim3(DMODEL / 256, M / 64), blk, 0, stream>>>(attnb, wob, out, M, DMODEL, DMODEL);
}
